// ComplexAttention_47356309406062
// MI455X (gfx1250) — matrix operand order repaired and checked by reading, not compiled, not run on hardware
//
#include <hip/hip_runtime.h>
#include <hip/hip_bf16.h>

typedef __attribute__((ext_vector_type(16))) _Float16 v16h;
typedef __attribute__((ext_vector_type(8)))  _Float16 v8h;
typedef __attribute__((ext_vector_type(8)))  float    v8f;

union AFrag { v16h v; v8h h[2]; };

#define WMMA_F16(a, b, c) \
  __builtin_amdgcn_wmma_f32_16x16x32_f16(false, (a), false, (b), (short)0, (c), false, false)

// ---------------------------------------------------------------------------
// Pack kernels
// ---------------------------------------------------------------------------

// Acat[m][k] (f16, 2048x2048): k<1024 -> x_real[m][k], else x_imag[m][k-1024]
__global__ void ca_pack_acat(const float* __restrict__ xr, const float* __restrict__ xi,
                             _Float16* __restrict__ Acat, int total) {
  int idx = blockIdx.x * 256 + threadIdx.x;
  if (idx >= total) return;
  int m = idx >> 11, k = idx & 2047;
  float v = (k < 1024) ? xr[m * 1024 + k] : xi[m * 1024 + (k - 1024)];
  Acat[idx] = (_Float16)v;
}

// Complexified weights, k-major per output row e:
//   TR[e][k] = k<1024 ?  Wr[e][k] : -Wi[e][k-1024]   (real-part GEMM operand)
//   TI[e][k] = k<1024 ?  Wi[e][k] :  Wr[e][k-1024]   (imag-part GEMM operand)
__global__ void ca_pack_w(const float* __restrict__ Wr, const float* __restrict__ Wi,
                          _Float16* __restrict__ TR, _Float16* __restrict__ TI, int total) {
  int idx = blockIdx.x * 256 + threadIdx.x;
  if (idx >= total) return;
  int e = idx >> 11, k = idx & 2047;
  float r, im;
  if (k < 1024) { float a = Wr[e * 1024 + k], b = Wi[e * 1024 + k]; r = a;  im = b; }
  else          { float a = Wr[e * 1024 + k - 1024], b = Wi[e * 1024 + k - 1024]; r = -b; im = a; }
  TR[idx] = (_Float16)r;
  TI[idx] = (_Float16)im;
}

// ---------------------------------------------------------------------------
// Complex GEMM via WMMA:  C = Acat @ B^T  (B stored k-contiguous per column e)
// 64x64 C tile per block: 4 waves x (4 M-subtiles x 16 cols).
// Each B fragment (bR,bI) is reused by 4 WMMAs -> 1.5 b128 loads per WMMA.
// mode 0: f16 outputs (qkv_r/qkv_i) ; mode 1: f32 output [2,B,N,D] with bias
// grid: (M/64, Nout/64), block 128
// ---------------------------------------------------------------------------
__global__ void ca_gemm_cplx(const _Float16* __restrict__ A,
                             const _Float16* __restrict__ BR,
                             const _Float16* __restrict__ BI,
                             const float* __restrict__ biasR,
                             const float* __restrict__ biasI,
                             _Float16* __restrict__ outRh, _Float16* __restrict__ outIh,
                             float* __restrict__ outF,
                             int K, int Nout, int imagOff, int mode) {
  int w = threadIdx.x >> 5, lane = threadIdx.x & 31;
  int hi = lane >> 4, ln = lane & 15;
  int etile = blockIdx.y * 4 + w;
  int m0 = blockIdx.x * 64;

  const _Float16* arow0 = A + (size_t)(m0 + ln) * K;          // subtile mt adds mt*16*K
  const _Float16* brrow = BR + (size_t)(etile * 16 + ln) * K;
  const _Float16* birow = BI + (size_t)(etile * 16 + ln) * K;

  v8f accR[4] = {};
  v8f accI[4] = {};
  int nk = K >> 5;
  for (int ks = 0; ks < nk; ++ks) {
    AFrag a[4];
#pragma unroll
    for (int mt = 0; mt < 4; ++mt) {
      const _Float16* ar = arow0 + (size_t)(mt * 16) * K;
      a[mt].h[0] = *(const v8h*)(ar + ks * 32 + hi * 16);       // K = ks*32 + hi*16 + [0..7]
      a[mt].h[1] = *(const v8h*)(ar + ks * 32 + hi * 16 + 8);  // K = ks*32 + hi*16 + 8 + [0..7]
    }
    v16h bR = *(const v16h*)(brrow + ks * 32 + hi * 16);       // col=ln, K = ks*32 + hi*16 + j
    v16h bI = *(const v16h*)(birow + ks * 32 + hi * 16);
#pragma unroll
    for (int mt = 0; mt < 4; ++mt) {
      accR[mt] = WMMA_F16(a[mt].v, bR, accR[mt]);
      accI[mt] = WMMA_F16(a[mt].v, bI, accI[mt]);
    }
  }

  int col = etile * 16 + ln;
  float br = biasR[col], bi = biasI[col];
#pragma unroll
  for (int mt = 0; mt < 4; ++mt) {
#pragma unroll
    for (int r = 0; r < 8; ++r) {
      int m = m0 + mt * 16 + r + hi * 8;       // C layout: M = r + hi*8, N = ln
      float vr = accR[mt][r] + br, vi = accI[mt][r] + bi;
      if (mode == 0) {
        outRh[(size_t)m * Nout + col] = (_Float16)vr;
        outIh[(size_t)m * Nout + col] = (_Float16)vi;
      } else {
        outF[(size_t)m * Nout + col] = vr;
        outF[(size_t)imagOff + (size_t)m * Nout + col] = vi;
      }
    }
  }
}

// ---------------------------------------------------------------------------
// Rotary + per-head repack.
// channel e = h*192 + d*3 + {0:q,1:k,2:v}  (qkv fastest in reshape)
// qcat/kcat:  [bh][n][0..63]=rot real, [64..127]=rot imag
// kcat2:      [bh][n][0..63]=-ki',     [64..127]=kr'    (conj(k) trick)
// vcatT:      [bh][c][n] with c: 0..63=vr, 64..127=vi   (k-contiguous B operand)
// ---------------------------------------------------------------------------
__global__ void ca_rotary_pack(const _Float16* __restrict__ qkvR, const _Float16* __restrict__ qkvI,
                               const float* __restrict__ fr_, const float* __restrict__ fi_,
                               _Float16* __restrict__ qcat, _Float16* __restrict__ kcat,
                               _Float16* __restrict__ kcat2, _Float16* __restrict__ vcatT,
                               int total) {
  int idx = blockIdx.x * 256 + threadIdx.x;
  if (idx >= total) return;
  int d = idx & 63;
  int n = (idx >> 6) & 1023;
  int h = (idx >> 16) & 15;
  int b = idx >> 20;
  int m = b * 1024 + n;
  int bh = b * 16 + h;
  int eb = h * 192 + d * 3;
  size_t qbase = (size_t)m * 3072 + eb;

  float qr = (float)qkvR[qbase + 0], qi = (float)qkvI[qbase + 0];
  float kr = (float)qkvR[qbase + 1], ki = (float)qkvI[qbase + 1];
  float vr = (float)qkvR[qbase + 2], vi = (float)qkvI[qbase + 2];
  float fr = fr_[n * 64 + d], fi = fi_[n * 64 + d];

  float qr2 = qr * fr - qi * fi, qi2 = qr * fi + qi * fr;
  float kr2 = kr * fr - ki * fi, ki2 = kr * fi + ki * fr;

  size_t row = ((size_t)(bh * 1024 + n)) * 128;
  qcat [row + d]       = (_Float16)qr2;
  qcat [row + 64 + d]  = (_Float16)qi2;
  kcat [row + d]       = (_Float16)kr2;
  kcat [row + 64 + d]  = (_Float16)ki2;
  kcat2[row + d]       = (_Float16)(-ki2);
  kcat2[row + 64 + d]  = (_Float16)kr2;

  size_t vb = (size_t)bh * 128 * 1024;
  vcatT[vb + (size_t)d * 1024 + n]        = (_Float16)vr;
  vcatT[vb + (size_t)(64 + d) * 1024 + n] = (_Float16)vi;
}

// ---------------------------------------------------------------------------
// Fused attention core: per (bh, 16-row query tile):
//   dots_{r,i} via WMMA (K'=128) -> |dots|*0.125 -> in-place LDS softmax ->
//   attn(f16) @ vcat via WMMA -> f16 outcat [m][2048] = [or | oi]
// block = 128 threads (4 waves); grid = B*H*(N/16) = 2048
// ---------------------------------------------------------------------------
__global__ void ca_attention(const _Float16* __restrict__ qcat,
                             const _Float16* __restrict__ kcat,
                             const _Float16* __restrict__ kcat2,
                             const _Float16* __restrict__ vcatT,
                             _Float16* __restrict__ outcat) {
  int bh = blockIdx.x >> 6;
  int ntile = blockIdx.x & 63;
  int b = bh >> 4, h = bh & 15;
  int tid = threadIdx.x, w = tid >> 5, lane = tid & 31;
  int hi = lane >> 4, ln = lane & 15;
  int n0 = ntile * 16;

  __shared__ _Float16 smag[16 * 1024];   // 32 KB score strip (in-place softmax)
  __shared__ float red[16][8];
  __shared__ float rowstat[16];

  // Q fragments for this 16-row tile (K'=128 -> 4 k-steps), reused for all 64 m-tiles
  const _Float16* qbase = qcat + ((size_t)(bh * 1024 + n0 + ln)) * 128;
  AFrag aq[4];
#pragma unroll
  for (int ks = 0; ks < 4; ++ks) {
    aq[ks].h[0] = *(const v8h*)(qbase + ks * 32 + hi * 16);
    aq[ks].h[1] = *(const v8h*)(qbase + ks * 32 + hi * 16 + 8);
  }

  // dots: wave w covers m-columns [w*256, w*256+255]
  for (int t = 0; t < 16; ++t) {
    int mtile = w * 16 + t;
    const _Float16* kb  = kcat  + ((size_t)(bh * 1024 + mtile * 16 + ln)) * 128;
    const _Float16* kb2 = kcat2 + ((size_t)(bh * 1024 + mtile * 16 + ln)) * 128;
    v8f accR = {}; v8f accI = {};
#pragma unroll
    for (int ks = 0; ks < 4; ++ks) {
      v16h bR = *(const v16h*)(kb  + ks * 32 + hi * 16);
      v16h bI = *(const v16h*)(kb2 + ks * 32 + hi * 16);
      accR = WMMA_F16(aq[ks].v, bR, accR);
      accI = WMMA_F16(aq[ks].v, bI, accI);
    }
#pragma unroll
    for (int r = 0; r < 8; ++r) {
      float dr = accR[r], di = accI[r];
      float mg = sqrtf(dr * dr + di * di) * 0.125f;   // HD^-0.5
      smag[(r + hi * 8) * 1024 + mtile * 16 + ln] = (_Float16)mg;
    }
  }
  __syncthreads();

  // softmax over 1024 cols per row; 8 threads/row, contiguous 128-col chunks
  int row = tid >> 3, sub = tid & 7;
  int base = row * 1024 + sub * 128;
  float mx = 0.0f;                                    // magnitudes are >= 0
  for (int c = 0; c < 128; ++c) mx = fmaxf(mx, (float)smag[base + c]);
  red[row][sub] = mx;
  __syncthreads();
  if (tid < 16) {
    float v = red[tid][0];
    for (int j = 1; j < 8; ++j) v = fmaxf(v, red[tid][j]);
    rowstat[tid] = v;
  }
  __syncthreads();
  float rmax = rowstat[row];
  float s = 0.0f;
  for (int c = 0; c < 128; ++c) {
    float e = __expf((float)smag[base + c] - rmax);
    smag[base + c] = (_Float16)e;
    s += e;
  }
  red[row][sub] = s;
  __syncthreads();
  if (tid < 16) {
    float v = 0.0f;
    for (int j = 0; j < 8; ++j) v += red[tid][j];
    rowstat[tid] = 1.0f / v;
  }
  __syncthreads();
  float inv = rowstat[row];
  for (int c = 0; c < 128; ++c)
    smag[base + c] = (_Float16)((float)smag[base + c] * inv);
  __syncthreads();

  // out = attn(16x1024, f16 in LDS) @ vcat(1024x128)
  // wave w owns ctiles {w, w+4}; one shared A fragment per k-step (3 loads / WMMA)
  int c0 = w * 16 + ln;
  int c1 = (w + 4) * 16 + ln;
  const _Float16* vb0 = vcatT + ((size_t)(bh * 128 + c0)) * 1024;
  const _Float16* vb1 = vcatT + ((size_t)(bh * 128 + c1)) * 1024;
  v8f acc0 = {}; v8f acc1 = {};
  for (int ks = 0; ks < 32; ++ks) {
    AFrag a;
    a.h[0] = *(const v8h*)(&smag[ln * 1024 + ks * 32 + hi * 16]);
    a.h[1] = *(const v8h*)(&smag[ln * 1024 + ks * 32 + hi * 16 + 8]);
    v16h b0 = *(const v16h*)(vb0 + ks * 32 + hi * 16);
    v16h b1 = *(const v16h*)(vb1 + ks * 32 + hi * 16);
    acc0 = WMMA_F16(a.v, b0, acc0);
    acc1 = WMMA_F16(a.v, b1, acc1);
  }
  // c<64 -> real component d=c ; c>=64 -> imag component d=c-64
  int co0 = (c0 < 64) ? (h * 64 + c0) : (1024 + h * 64 + (c0 - 64));
  int co1 = (c1 < 64) ? (h * 64 + c1) : (1024 + h * 64 + (c1 - 64));
#pragma unroll
  for (int r = 0; r < 8; ++r) {
    int gm = b * 1024 + n0 + r + hi * 8;
    outcat[(size_t)gm * 2048 + co0] = (_Float16)acc0[r];
    outcat[(size_t)gm * 2048 + co1] = (_Float16)acc1[r];
  }
}

// ---------------------------------------------------------------------------
extern "C" void kernel_launch(void* const* d_in, const int* in_sizes, int n_in,
                              void* d_out, int out_size, void* d_ws, size_t ws_size,
                              hipStream_t stream) {
  const float* x_real  = (const float*)d_in[0];
  const float* x_imag  = (const float*)d_in[1];
  const float* freqs_r = (const float*)d_in[2];
  const float* freqs_i = (const float*)d_in[3];
  const float* Wqkv_r  = (const float*)d_in[4];
  const float* Wqkv_i  = (const float*)d_in[5];
  const float* bqkv_r  = (const float*)d_in[6];
  const float* bqkv_i  = (const float*)d_in[7];
  const float* Wout_r  = (const float*)d_in[8];
  const float* Wout_i  = (const float*)d_in[9];
  const float* bout_r  = (const float*)d_in[10];
  const float* bout_i  = (const float*)d_in[11];
  float* out = (float*)d_out;

  char* ws = (char*)d_ws;
  size_t off = 0;
  auto take = [&](size_t bytes) -> char* {
    char* p = ws + off;
    off += (bytes + 255) & ~(size_t)255;
    return p;
  };

  _Float16* Acat    = (_Float16*)take((size_t)2048 * 2048 * 2);
  _Float16* WqkvRT  = (_Float16*)take((size_t)3072 * 2048 * 2);
  _Float16* WqkvIT  = (_Float16*)take((size_t)3072 * 2048 * 2);
  _Float16* WoutRT  = (_Float16*)take((size_t)1024 * 2048 * 2);
  _Float16* WoutIT  = (_Float16*)take((size_t)1024 * 2048 * 2);
  _Float16* qkvR    = (_Float16*)take((size_t)2048 * 3072 * 2);
  _Float16* qkvI    = (_Float16*)take((size_t)2048 * 3072 * 2);
  _Float16* qcat    = (_Float16*)take((size_t)32 * 1024 * 128 * 2);
  _Float16* kcat    = (_Float16*)take((size_t)32 * 1024 * 128 * 2);
  _Float16* kcat2   = (_Float16*)take((size_t)32 * 1024 * 128 * 2);
  _Float16* vcatT   = (_Float16*)take((size_t)32 * 128 * 1024 * 2);
  _Float16* outcat  = (_Float16*)take((size_t)2048 * 2048 * 2);

  // 1) pack inputs/weights to f16 complexified operands
  {
    int total = 2048 * 2048;
    ca_pack_acat<<<(total + 255) / 256, 256, 0, stream>>>(x_real, x_imag, Acat, total);
  }
  {
    int total = 3072 * 2048;
    ca_pack_w<<<(total + 255) / 256, 256, 0, stream>>>(Wqkv_r, Wqkv_i, WqkvRT, WqkvIT, total);
  }
  {
    int total = 1024 * 2048;
    ca_pack_w<<<(total + 255) / 256, 256, 0, stream>>>(Wout_r, Wout_i, WoutRT, WoutIT, total);
  }

  // 2) qkv = x @ Wqkv^T + bqkv   (M=2048, K'=2048, N=3072), f16 out, 64x64 tiles
  ca_gemm_cplx<<<dim3(2048 / 64, 3072 / 64), 128, 0, stream>>>(
      Acat, WqkvRT, WqkvIT, bqkv_r, bqkv_i, qkvR, qkvI, nullptr,
      2048, 3072, 0, 0);

  // 3) rotary + head repack
  {
    int total = 2 * 16 * 1024 * 64;
    ca_rotary_pack<<<(total + 255) / 256, 256, 0, stream>>>(
        qkvR, qkvI, freqs_r, freqs_i, qcat, kcat, kcat2, vcatT, total);
  }

  // 4) fused attention -> outcat f16 [2048][2048] = [or | oi]
  ca_attention<<<2 * 16 * 64, 128, 0, stream>>>(qcat, kcat, kcat2, vcatT, outcat);

  // 5) final projection -> f32 d_out [2, B, N, D], 64x64 tiles
  ca_gemm_cplx<<<dim3(2048 / 64, 1024 / 64), 128, 0, stream>>>(
      outcat, WoutRT, WoutIT, bout_r, bout_i, nullptr, nullptr, out,
      2048, 1024, 2 * 1024 * 1024, 1);
}
